// MultiHeadAttention_58299886076571
// MI455X (gfx1250) — matrix operand order repaired and checked by reading, not compiled, not run on hardware
//
#include <hip/hip_runtime.h>

// ---------------------------------------------------------------------------
// MI455X (gfx1250) multi-head attention, bf16 WMMA path with f32 accumulate.
//   - wave32, V_WMMA_F32_16X16X32_BF16 everywhere (K=32 per instruction)
//   - LDS-tiled GEMMs + flash-attention with online softmax
//   - async global->LDS staging (ASYNCcnt) where no transpose is needed
// ---------------------------------------------------------------------------

typedef __bf16 bf16;
typedef __attribute__((ext_vector_type(16))) __bf16 bf16x16;
typedef __attribute__((ext_vector_type(8)))  __bf16 bf16x8;
typedef __attribute__((ext_vector_type(4)))  __bf16 bf16x4;
typedef __attribute__((ext_vector_type(8)))  float  f32x8;
typedef int v4i __attribute__((vector_size(16)));  // matches builtin param type

#define AS1 __attribute__((address_space(1)))
#define AS3 __attribute__((address_space(3)))

#if defined(__has_builtin)
#if __has_builtin(__builtin_amdgcn_global_load_async_to_lds_b128) && \
    __has_builtin(__builtin_amdgcn_s_wait_asynccnt)
#define HAVE_ASYNC_LDS 1
#endif
#endif
#ifndef HAVE_ASYNC_LDS
#define HAVE_ASYNC_LDS 0
#endif

__device__ __forceinline__ void async_copy_b128(const void* g, void* l) {
#if HAVE_ASYNC_LDS
  // Generic->AS casts via integers: LDS generic pointers carry the LDS offset
  // in the low 32 bits; global generic pointers are bit-identical to AS1.
  __builtin_amdgcn_global_load_async_to_lds_b128(
      (AS1 v4i*)(unsigned long long)g,
      (AS3 v4i*)(unsigned int)(unsigned long long)l, 0, 0);
#else
  *(bf16x8*)l = *(const bf16x8*)g;
#endif
}
__device__ __forceinline__ void async_wait0() {
#if HAVE_ASYNC_LDS
  __builtin_amdgcn_s_wait_asynccnt(0);
#endif
}

__device__ __forceinline__ bf16x16 join16(bf16x8 lo, bf16x8 hi) {
  return __builtin_shufflevector(lo, hi, 0,1,2,3,4,5,6,7,8,9,10,11,12,13,14,15);
}

__device__ __forceinline__ f32x8 wmma_bf16(bf16x16 a, bf16x16 b, f32x8 c) {
  // (neg_a, A, neg_b, B, c_mod, C, reuse_a, reuse_b)
  return __builtin_amdgcn_wmma_f32_16x16x32_bf16(false, a, false, b, (short)0, c,
                                                 false, false);
}

// ------------------------------ fp32 -> bf16 -------------------------------
__global__ __launch_bounds__(256) void cvt_f32_to_bf16(
    const float* __restrict__ in, bf16* __restrict__ out, int n) {
  int i = (blockIdx.x * blockDim.x + threadIdx.x) * 4;
  if (i + 3 < n) {
    float4 v = *(const float4*)(in + i);
    bf16x4 o;
    o[0] = (bf16)v.x; o[1] = (bf16)v.y; o[2] = (bf16)v.z; o[3] = (bf16)v.w;
    *(bf16x4*)(out + i) = o;
  }
}

// ------------------------------ WMMA GEMM ----------------------------------
// C[M,N] = A[M,K] * W[K,N] + bias[N]
// MODE 0: out = float, row-major [M][N]
// MODE 1: out = bf16 in head layout [B,H,S,hd] with n = h*64+d, m = b*S+s
template <int MODE>
__global__ __launch_bounds__(256) void gemm_bf16_128x128(
    const bf16* __restrict__ A, const bf16* __restrict__ W,
    const float* __restrict__ bias, void* __restrict__ outp,
    int M, int N, int K, int S, int H) {
  constexpr int BM = 128, BN = 128, KT = 64;
  __shared__ __align__(16) bf16 As[BM * KT];   // [m][k]  16 KB
  __shared__ __align__(16) bf16 Bs[BN * KT];   // [n][k]  16 KB (transposed)

  const int tid  = threadIdx.x;
  const int lane = tid & 31;
  const int wid  = tid >> 5;       // 0..7
  const int wm   = wid & 3;        // 4 wave-rows of 32
  const int wn   = wid >> 2;       // 2 wave-cols of 64
  const int half = lane >> 4;      // 0/1
  const int ln16 = lane & 15;

  const int m0 = blockIdx.y * BM;
  const int n0 = blockIdx.x * BN;

  f32x8 acc[2][4];
#pragma unroll
  for (int mi = 0; mi < 2; ++mi)
#pragma unroll
    for (int ni = 0; ni < 4; ++ni)
#pragma unroll
      for (int r = 0; r < 8; ++r) acc[mi][ni][r] = 0.0f;

  const int kTiles = K / KT;
  for (int kt = 0; kt < kTiles; ++kt) {
    const int k0 = kt * KT;
    // Stage A tile: 128x64 bf16, 32 contiguous elems/thread, async -> LDS.
    {
      int row  = tid >> 1;
      int col0 = (tid & 1) * 32;
      const bf16* g = A + (size_t)(m0 + row) * K + k0 + col0;
      bf16* s = &As[row * KT + col0];
#pragma unroll
      for (int i = 0; i < 4; ++i) async_copy_b128(g + i * 8, s + i * 8);
    }
    // Stage B tile transposed: W[k][n] -> Bs[n][k] (VGPR scatter path).
#pragma unroll
    for (int s2 = 0; s2 < 4; ++s2) {
      int seg = tid * 4 + s2;       // 0..1023
      int kk  = seg >> 4;           // 0..63
      int nn  = (seg & 15) * 8;     // 0..120
      bf16x8 v = *(const bf16x8*)(W + (size_t)(k0 + kk) * N + n0 + nn);
#pragma unroll
      for (int j = 0; j < 8; ++j) Bs[(nn + j) * KT + kk] = v[j];
    }
    if (kt + 1 < kTiles) {  // pull next tiles toward L2/WGP$ early
      __builtin_prefetch(A + (size_t)(m0 + (tid >> 1)) * K + k0 + KT, 0, 3);
      __builtin_prefetch(W + (size_t)(k0 + KT + (tid & 63)) * N + n0, 0, 3);
    }
    async_wait0();
    __syncthreads();

#pragma unroll
    for (int ks = 0; ks < 2; ++ks) {
      bf16x16 af[2];
#pragma unroll
      for (int mi = 0; mi < 2; ++mi) {
        const bf16* base = &As[(wm * 32 + mi * 16 + ln16) * KT + ks * 32];
        af[mi] = join16(*(const bf16x8*)(base + half * 8),
                        *(const bf16x8*)(base + 16 + half * 8));
      }
      bf16x16 bf[4];
#pragma unroll
      for (int ni = 0; ni < 4; ++ni) {
        const bf16* base = &Bs[(wn * 64 + ni * 16 + ln16) * KT + ks * 32 + half * 8];
        bf[ni] = join16(*(const bf16x8*)base, *(const bf16x8*)(base + 16));
      }
#pragma unroll
      for (int mi = 0; mi < 2; ++mi)
#pragma unroll
        for (int ni = 0; ni < 4; ++ni)
          acc[mi][ni] = wmma_bf16(af[mi], bf[ni], acc[mi][ni]);
    }
    __syncthreads();
  }

  // Epilogue: C/D layout -> element r of VGPR = row (wm*32+mi*16+half*8+r), col = lane%16
#pragma unroll
  for (int mi = 0; mi < 2; ++mi)
#pragma unroll
    for (int ni = 0; ni < 4; ++ni) {
      int n = n0 + wn * 64 + ni * 16 + ln16;
      float bv = bias[n];
#pragma unroll
      for (int r = 0; r < 8; ++r) {
        float val = acc[mi][ni][r] + bv;
        int m = m0 + wm * 32 + mi * 16 + half * 8 + r;
        if (MODE == 0) {
          ((float*)outp)[(size_t)m * N + n] = val;
        } else {
          int b = m / S, s = m % S;
          int h = n >> 6, d = n & 63;
          ((bf16*)outp)[(((size_t)(b * H + h) * S + s) << 6) + d] = (bf16)val;
        }
      }
    }
}

// --------------------------- flash attention -------------------------------
// One block per (b, h, 128-query tile). 8 waves; wave w owns 16 query rows.
__global__ __launch_bounds__(256) void attn_flash(
    const bf16* __restrict__ Q, const bf16* __restrict__ Km,
    const bf16* __restrict__ V, bf16* __restrict__ ctx, int B, int H, int S) {
  constexpr int hd = 64, BQ = 128, BK = 64;
  __shared__ __align__(16) bf16 Qs[BQ * hd];      // 16 KB [q][d]
  __shared__ __align__(16) bf16 Ks[BK * hd];      //  8 KB [kk][d]  (= B layout [N][K])
  __shared__ __align__(16) bf16 VTs[hd * BK];     //  8 KB [d][kk]  (= B layout [N][K])
  __shared__ __align__(16) bf16 Ps[8 * 16 * BK];  // 16 KB per-wave P [16][64]

  const int tid  = threadIdx.x;
  const int lane = tid & 31;
  const int wid  = tid >> 5;
  const int half = lane >> 4;
  const int ln16 = lane & 15;

  const int q0 = blockIdx.x * BQ;
  const int h  = blockIdx.y;
  const int b  = blockIdx.z;
  const size_t headOff = ((size_t)(b * H + h)) * S * hd;

  {  // stage Q tile (contiguous in [B,H,S,hd]) -> async to LDS
    const bf16* g = Q + headOff + (size_t)q0 * hd + tid * 32;
    bf16* s = Qs + tid * 32;
#pragma unroll
    for (int i = 0; i < 4; ++i) async_copy_b128(g + i * 8, s + i * 8);
  }

  f32x8 accO[4], m_run, l_run;
#pragma unroll
  for (int r = 0; r < 8; ++r) { m_run[r] = -1e30f; l_run[r] = 0.0f; }
#pragma unroll
  for (int ni = 0; ni < 4; ++ni)
#pragma unroll
    for (int r = 0; r < 8; ++r) accO[ni][r] = 0.0f;

  const int jEnd = (q0 + BQ - 1) / BK;  // causal upper key block (inclusive)
  for (int j = 0; j <= jEnd; ++j) {
    const int kb = j * BK;
    __syncthreads();  // protect Ks/VTs before overwrite
    {  // stage K block (row-major copy; already in B-fragment layout) -> async
      const bf16* g = Km + headOff + (size_t)kb * hd + tid * 16;
      bf16* s = Ks + tid * 16;
      async_copy_b128(g, s);
      async_copy_b128(g + 8, s + 8);
    }
#pragma unroll
    for (int s2 = 0; s2 < 2; ++s2) {  // stage V transposed -> [d][kk]
      int seg = tid * 2 + s2;
      int kk  = seg >> 3;
      int d0  = (seg & 7) * 8;
      bf16x8 v = *(const bf16x8*)(V + headOff + (size_t)(kb + kk) * hd + d0);
#pragma unroll
      for (int jj = 0; jj < 8; ++jj) VTs[(d0 + jj) * BK + kk] = v[jj];
    }
    if (j < jEnd) {
      __builtin_prefetch(Km + headOff + (size_t)(kb + BK) * hd + tid * 16, 0, 3);
      __builtin_prefetch(V  + headOff + (size_t)(kb + BK) * hd + tid * 16, 0, 3);
    }
    async_wait0();  // Q (first iter) + K async stores must be LDS-visible
    __syncthreads();

    // ---- scores: S = Q * K^T  (16q x 64k per wave) ----
    f32x8 sc[4];
#pragma unroll
    for (int ni = 0; ni < 4; ++ni)
#pragma unroll
      for (int r = 0; r < 8; ++r) sc[ni][r] = 0.0f;
#pragma unroll
    for (int ks = 0; ks < 2; ++ks) {
      const bf16* qb = &Qs[(wid * 16 + ln16) * hd + ks * 32];
      bf16x16 a = join16(*(const bf16x8*)(qb + half * 8),
                         *(const bf16x8*)(qb + 16 + half * 8));
#pragma unroll
      for (int ni = 0; ni < 4; ++ni) {
        const bf16* kbse = &Ks[(ni * 16 + ln16) * hd + ks * 32 + half * 8];
        bf16x16 bb = join16(*(const bf16x8*)kbse, *(const bf16x8*)(kbse + 16));
        sc[ni] = wmma_bf16(a, bb, sc[ni]);
      }
    }

    // ---- scale + causal mask (select, no divergence) ----
    const float scale = 0.125f;  // 1/sqrt(64)
#pragma unroll
    for (int ni = 0; ni < 4; ++ni) {
      int kkIdx = kb + ni * 16 + ln16;
#pragma unroll
      for (int r = 0; r < 8; ++r) {
        int q = q0 + wid * 16 + half * 8 + r;
        float v = sc[ni][r] * scale;
        sc[ni][r] = (kkIdx > q) ? -1e30f : v;
      }
    }

    // ---- online softmax (rows striped over 16-lane halves) ----
    f32x8 mb;
#pragma unroll
    for (int r = 0; r < 8; ++r)
      mb[r] = fmaxf(fmaxf(sc[0][r], sc[1][r]), fmaxf(sc[2][r], sc[3][r]));
#pragma unroll
    for (int off = 1; off < 16; off <<= 1)
#pragma unroll
      for (int r = 0; r < 8; ++r)
        mb[r] = fmaxf(mb[r], __shfl_xor(mb[r], off, 32));

    f32x8 m_new, alpha, rs;
#pragma unroll
    for (int r = 0; r < 8; ++r) {
      m_new[r] = fmaxf(m_run[r], mb[r]);
      alpha[r] = __expf(m_run[r] - m_new[r]);
      rs[r] = 0.0f;
    }
#pragma unroll
    for (int ni = 0; ni < 4; ++ni)
#pragma unroll
      for (int r = 0; r < 8; ++r) {
        float p = __expf(sc[ni][r] - m_new[r]);
        sc[ni][r] = p;
        rs[r] += p;
      }
#pragma unroll
    for (int off = 1; off < 16; off <<= 1)
#pragma unroll
      for (int r = 0; r < 8; ++r) rs[r] += __shfl_xor(rs[r], off, 32);
#pragma unroll
    for (int r = 0; r < 8; ++r) {
      l_run[r] = l_run[r] * alpha[r] + rs[r];
      m_run[r] = m_new[r];
    }
#pragma unroll
    for (int ni = 0; ni < 4; ++ni)
#pragma unroll
      for (int r = 0; r < 8; ++r) accO[ni][r] *= alpha[r];

    // ---- C/D -> A-fragment relayout of P through per-wave LDS ----
    bf16* Pw = &Ps[wid * 16 * BK];
#pragma unroll
    for (int ni = 0; ni < 4; ++ni)
#pragma unroll
      for (int r = 0; r < 8; ++r)
        Pw[(half * 8 + r) * BK + ni * 16 + ln16] = (bf16)sc[ni][r];
    asm volatile("" ::: "memory");  // same-wave LDS is in-order; stop reordering

    // ---- O += P * V ----
#pragma unroll
    for (int ks = 0; ks < 2; ++ks) {
      const bf16* pb = &Pw[ln16 * BK + ks * 32];
      bf16x16 ap = join16(*(const bf16x8*)(pb + half * 8),
                          *(const bf16x8*)(pb + 16 + half * 8));
#pragma unroll
      for (int ni = 0; ni < 4; ++ni) {
        const bf16* vb = &VTs[(ni * 16 + ln16) * BK + ks * 32 + half * 8];
        bf16x16 bv = join16(*(const bf16x8*)vb, *(const bf16x8*)(vb + 16));
        accO[ni] = wmma_bf16(ap, bv, accO[ni]);
      }
    }
  }

  // ---- normalize + store ctx as bf16 [B,S,D] (D = H*hd) ----
  const int D = H * hd;
#pragma unroll
  for (int ni = 0; ni < 4; ++ni) {
    int d = ni * 16 + ln16;
#pragma unroll
    for (int r = 0; r < 8; ++r) {
      int q = q0 + wid * 16 + half * 8 + r;
      float o = accO[ni][r] / l_run[r];
      ctx[((size_t)b * S + q) * D + h * hd + d] = (bf16)o;
    }
  }
}

// ------------------------------- launcher ----------------------------------
extern "C" void kernel_launch(void* const* d_in, const int* in_sizes, int n_in,
                              void* d_out, int out_size, void* d_ws, size_t ws_size,
                              hipStream_t stream) {
  const int B = 2, S = 2048, D = 1024, H = 16, hd = 64;
  const int M = B * S;  // 4096
  (void)hd;

  const float* x  = (const float*)d_in[0];
  const float* Wq = (const float*)d_in[1];
  const float* bq = (const float*)d_in[2];
  const float* Wk = (const float*)d_in[3];
  const float* bk = (const float*)d_in[4];
  const float* Wv = (const float*)d_in[5];
  const float* bv = (const float*)d_in[6];
  const float* Wo = (const float*)d_in[7];
  const float* bo = (const float*)d_in[8];

  char* p = (char*)d_ws;
  bf16* xb   = (bf16*)p;                 p += (size_t)M * D * 2;  // 8 MiB
  bf16* wqb  = (bf16*)p;                 p += (size_t)D * D * 2;  // 2 MiB
  bf16* wkb  = (bf16*)p;                 p += (size_t)D * D * 2;
  bf16* wvb  = (bf16*)p;                 p += (size_t)D * D * 2;
  bf16* wob  = (bf16*)p;                 p += (size_t)D * D * 2;
  bf16* Qb   = (bf16*)p;                 p += (size_t)M * D * 2;  // [B,H,S,hd]
  bf16* Kb   = (bf16*)p;                 p += (size_t)M * D * 2;
  bf16* Vb   = (bf16*)p;                 p += (size_t)M * D * 2;
  bf16* ctxb = (bf16*)p;                 p += (size_t)M * D * 2;  // [B,S,D]

  // 1) fp32 -> bf16
  cvt_f32_to_bf16<<<(M * D) / 1024, 256, 0, stream>>>(x, xb, M * D);
  cvt_f32_to_bf16<<<(D * D) / 1024, 256, 0, stream>>>(Wq, wqb, D * D);
  cvt_f32_to_bf16<<<(D * D) / 1024, 256, 0, stream>>>(Wk, wkb, D * D);
  cvt_f32_to_bf16<<<(D * D) / 1024, 256, 0, stream>>>(Wv, wvb, D * D);
  cvt_f32_to_bf16<<<(D * D) / 1024, 256, 0, stream>>>(Wo, wob, D * D);

  // 2) Q/K/V projections (bf16 WMMA, bf16 head-layout output)
  dim3 gg(D / 128, M / 128);
  gemm_bf16_128x128<1><<<gg, 256, 0, stream>>>(xb, wqb, bq, Qb, M, D, D, S, H);
  gemm_bf16_128x128<1><<<gg, 256, 0, stream>>>(xb, wkb, bk, Kb, M, D, D, S, H);
  gemm_bf16_128x128<1><<<gg, 256, 0, stream>>>(xb, wvb, bv, Vb, M, D, D, S, H);

  // 3) causal flash attention
  dim3 ga(S / 128, H, B);
  attn_flash<<<ga, 256, 0, stream>>>(Qb, Kb, Vb, ctxb, B, H, S);

  // 4) output projection, fp32 result straight into d_out
  gemm_bf16_128x128<0><<<gg, 256, 0, stream>>>(ctxb, wob, bo, d_out, M, D, D, S, H);
}
